// DEAM_60842506715328
// MI455X (gfx1250) — compile-verified, matrix operand order checked by reading, not run on hardware
//
#include <hip/hip_runtime.h>
#include <cstdint>
#include <cmath>

// Problem constants (from reference)
#define D_    240
#define KP_   256               // padded K-stride for D=240 operands
#define NH_   8
#define DH_   30
#define DFF_  960
#define NL_   3
#define NP_   4
#define B_    8
#define LEN_  5376              // 64*64 + 32*32 + 16*16
#define ROWS_ (B_ * LEN_)       // 43008 (multiple of 64)

typedef __attribute__((ext_vector_type(16))) __bf16 v16bf;
typedef __attribute__((ext_vector_type(8)))  float  v8f;

#if __has_builtin(__builtin_amdgcn_global_load_async_to_lds_b128) && \
    __has_builtin(__builtin_amdgcn_s_wait_asynccnt)
#define ASYNC_STAGE 1
typedef int v4i_vec __attribute__((vector_size(16)));
typedef __attribute__((address_space(1))) v4i_vec* gv4i_p;   // global (printed __device__)
typedef __attribute__((address_space(3))) v4i_vec* lv4i_p;   // LDS
#define GLOAD_ASYNC_B128(gptr, lptr)                                   \
  __builtin_amdgcn_global_load_async_to_lds_b128(                      \
      (gv4i_p)(gptr), (lv4i_p)(lptr), 0, 0)
#else
#define ASYNC_STAGE 0
#endif

__device__ __forceinline__ unsigned short f2bf(float f) {
  unsigned u = __float_as_uint(f);
  u += 0x7FFFu + ((u >> 16) & 1u);          // round-to-nearest-even
  return (unsigned short)(u >> 16);
}

union FragBF { uint4 q[2]; v16bf v; };
union AccF   { v8f v; float f[8]; };

// ---------------------------------------------------------------------------
// bf16 WMMA GEMM:  C[M,N] = A[M,Kp] * Wt[N,Kp]^T   (both row stride = Kp)
//  - M multiple of 64, Kp multiple of 32, Wt rows padded to next mult of 64.
//  - A pad columns are zero (writers guarantee), Wt pad region zeroed.
//  - Staging via async global->LDS (double buffered) when available,
//    else unguarded vectorized b128 loads + ds_store_b128.
// epi: 0=none  1=+bias  2=+bias,relu  3=*scale+bias,relu (conv BN epilogue)
// Output row remap: orow = (row/seg)*ostride + ooff + row%seg  (conv -> src)
// ---------------------------------------------------------------------------
__global__ __launch_bounds__(128)
void gemm_bf16_wmma(const unsigned short* __restrict__ A,
                    const unsigned short* __restrict__ Wt,
                    int M, int N, int Kp,
                    float* __restrict__ outF,
                    unsigned short* __restrict__ outU,
                    const float* __restrict__ bias,
                    const float* __restrict__ scale,
                    int epi, int seg, int ostride, int ooff) {
  __shared__ unsigned short As[2][64][40];   // [buf][m][k] row = 80B (16B aligned)
  __shared__ unsigned short Bs[2][64][40];   // [buf][n][k]

  const int t      = threadIdx.x;
  const int lane   = t & 31;
  const int wave   = t >> 5;
  const int waveM  = wave >> 1, waveN = wave & 1;
  const int m0     = blockIdx.y * 64;
  const int n0     = blockIdx.x * 64;
  const int laneLo = lane & 15;
  const int laneHi = lane >> 4;
  const int ka     = laneHi * 8;          // A-frag halves 0-7:  K = laneHi*8 + r
  const int kb     = 16 + laneHi * 8;     //        halves 8-15: K = 16 + laneHi*8 + r

  // Staging map: thread t -> row t>>1, 16-half chunk (t&1)*16
  const int lrow = t >> 1;
  const int lk   = (t & 1) * 16;
  const unsigned short* Ag = A  + (size_t)(m0 + lrow) * Kp + lk;
  const unsigned short* Bg = Wt + (size_t)(n0 + lrow) * Kp + lk;

  AccF acc[2][2];
#pragma unroll
  for (int i = 0; i < 2; ++i)
#pragma unroll
    for (int j = 0; j < 2; ++j)
#pragma unroll
      for (int r = 0; r < 8; ++r) acc[i][j].f[r] = 0.f;

  const int iters = Kp >> 5;

#if ASYNC_STAGE
  // Prologue: async-stage tile 0 into buffer 0 (ASYNCcnt path)
  GLOAD_ASYNC_B128(Ag,     &As[0][lrow][lk]);
  GLOAD_ASYNC_B128(Ag + 8, &As[0][lrow][lk + 8]);
  GLOAD_ASYNC_B128(Bg,     &Bs[0][lrow][lk]);
  GLOAD_ASYNC_B128(Bg + 8, &Bs[0][lrow][lk + 8]);
  for (int it = 0; it < iters; ++it) {
    const int buf = it & 1;
    if (it + 1 < iters) {
      const int kn = (it + 1) << 5;
      const int nb = (it + 1) & 1;
      GLOAD_ASYNC_B128(Ag + kn,     &As[nb][lrow][lk]);
      GLOAD_ASYNC_B128(Ag + kn + 8, &As[nb][lrow][lk + 8]);
      GLOAD_ASYNC_B128(Bg + kn,     &Bs[nb][lrow][lk]);
      GLOAD_ASYNC_B128(Bg + kn + 8, &Bs[nb][lrow][lk + 8]);
      __builtin_amdgcn_s_wait_asynccnt(4);   // current tile's 4 ops done
    } else {
      __builtin_amdgcn_s_wait_asynccnt(0);
    }
    __syncthreads();                         // all waves' tile data visible

    FragBF a[2], b[2];
#pragma unroll
    for (int i = 0; i < 2; ++i) {
      int row = waveM * 32 + i * 16 + laneLo;
      a[i].q[0] = *(const uint4*)&As[buf][row][ka];
      a[i].q[1] = *(const uint4*)&As[buf][row][kb];
      int col = waveN * 32 + i * 16 + laneLo;
      b[i].q[0] = *(const uint4*)&Bs[buf][col][ka];
      b[i].q[1] = *(const uint4*)&Bs[buf][col][kb];
    }
#pragma unroll
    for (int i = 0; i < 2; ++i)
#pragma unroll
      for (int j = 0; j < 2; ++j)
        acc[i][j].v = __builtin_amdgcn_wmma_f32_16x16x32_bf16(
            false, a[i].v, false, b[j].v, (short)0, acc[i][j].v, false, false);
    __syncthreads();                         // done reading buf before reuse
  }
#else
  for (int it = 0; it < iters; ++it) {
    const int k0 = it << 5;
    uint4 a0 = *(const uint4*)(Ag + k0);
    uint4 a1 = *(const uint4*)(Ag + k0 + 8);
    uint4 b0 = *(const uint4*)(Bg + k0);
    uint4 b1 = *(const uint4*)(Bg + k0 + 8);
    if (k0 + 32 < Kp) {                      // gfx1250 global_prefetch_b8
      __builtin_prefetch(Ag + k0 + 32, 0, 1);
      __builtin_prefetch(Bg + k0 + 32, 0, 1);
    }
    __syncthreads();
    *(uint4*)&As[0][lrow][lk]     = a0;
    *(uint4*)&As[0][lrow][lk + 8] = a1;
    *(uint4*)&Bs[0][lrow][lk]     = b0;
    *(uint4*)&Bs[0][lrow][lk + 8] = b1;
    __syncthreads();

    FragBF a[2], b[2];
#pragma unroll
    for (int i = 0; i < 2; ++i) {
      int row = waveM * 32 + i * 16 + laneLo;
      a[i].q[0] = *(const uint4*)&As[0][row][ka];
      a[i].q[1] = *(const uint4*)&As[0][row][kb];
      int col = waveN * 32 + i * 16 + laneLo;
      b[i].q[0] = *(const uint4*)&Bs[0][col][ka];
      b[i].q[1] = *(const uint4*)&Bs[0][col][kb];
    }
#pragma unroll
    for (int i = 0; i < 2; ++i)
#pragma unroll
      for (int j = 0; j < 2; ++j)
        acc[i][j].v = __builtin_amdgcn_wmma_f32_16x16x32_bf16(
            false, a[i].v, false, b[j].v, (short)0, acc[i][j].v, false, false);
  }
#endif

  // Store: C/D layout — lane<16: N=lane, VGPR r -> M=r; lane>=16: M=8+r
#pragma unroll
  for (int i = 0; i < 2; ++i)
#pragma unroll
    for (int j = 0; j < 2; ++j) {
      int gcol = n0 + waveN * 32 + j * 16 + laneLo;
#pragma unroll
      for (int r = 0; r < 8; ++r) {
        int grow = m0 + waveM * 32 + i * 16 + laneHi * 8 + r;
        if (gcol < N) {
          float v = acc[i][j].f[r];
          if (epi == 1 || epi == 2)      v += bias[gcol];
          else if (epi == 3)             v = v * scale[gcol] + bias[gcol];
          if (epi >= 2)                  v = fmaxf(v, 0.f);
          size_t orow = (size_t)(grow / seg) * ostride + ooff + (grow % seg);
          size_t oi = orow * (size_t)N + gcol;
          if (outF) outF[oi] = v;
          if (outU) outU[oi] = f2bf(v);
        }
      }
    }
}

// ---------------------------------------------------------------------------
// im2col (3x3, pad 1) -> bf16 A matrix [B*H*W, cin*9]   (cin*9 mult of 32)
// ---------------------------------------------------------------------------
__global__ void im2col_k(const float* __restrict__ x, unsigned short* __restrict__ A,
                         int cin, int H, int W, long long total) {
  long long idx = (long long)blockIdx.x * blockDim.x + threadIdx.x;
  if (idx >= total) return;
  int K = cin * 9;
  long long row = idx / K;
  int k = (int)(idx - row * K);
  int HW = H * W;
  int b = (int)(row / HW);
  int hw = (int)(row - (long long)b * HW);
  int y = hw / W, xq = hw - y * W;
  int ci = k / 9, r9 = k - ci * 9;
  int kh = r9 / 3, kw = r9 - kh * 3;
  int yy = y + kh - 1, xx = xq + kw - 1;
  float v = 0.f;
  if (yy >= 0 && yy < H && xx >= 0 && xx < W)
    v = x[(((size_t)b * cin + ci) * H + yy) * W + xx];
  A[idx] = f2bf(v);
}

// Conv weight OIHW -> bf16 Wt [Np=256][K]  (rows >= 240 zeroed)
__global__ void convw_k(const float* __restrict__ w, unsigned short* __restrict__ Wt,
                        int cin, int total) {
  int idx = blockIdx.x * blockDim.x + threadIdx.x;
  if (idx >= total) return;
  int K = cin * 9;
  int n = idx / K, k = idx - n * K;     // n = co
  float v = 0.f;
  if (n < D_) {
    int ci = k / 9, r9 = k - ci * 9;
    v = w[((size_t)n * cin + ci) * 9 + r9];
  }
  Wt[idx] = f2bf(v);
}

// Linear weight [K][N] f32 -> transposed bf16 Wt [Np][Kp] (pad zeroed)
__global__ void cvtT_k(const float* __restrict__ in, unsigned short* __restrict__ Wt,
                       int K, int N, int Kp, int total) {
  int idx = blockIdx.x * blockDim.x + threadIdx.x;
  if (idx >= total) return;
  int n = idx / Kp, k = idx - n * Kp;
  float v = (n < N && k < K) ? in[(size_t)k * N + n] : 0.f;
  Wt[idx] = f2bf(v);
}

// ---------------------------------------------------------------------------
// Sinusoidal pos embedding + level embed; q = src + pos (bf16, stride 256)
// ---------------------------------------------------------------------------
__global__ void posq_k(const float* __restrict__ src, const float* __restrict__ lvl_emb,
                       unsigned short* __restrict__ qU, unsigned short* __restrict__ srcU) {
  long long idx = (long long)blockIdx.x * blockDim.x + threadIdx.x;
  long long total = (long long)ROWS_ * KP_;
  if (idx >= total) return;
  int c = (int)(idx % KP_);
  long long row = idx / KP_;
  if (c >= D_) { qU[idx] = 0; srcU[idx] = 0; return; }   // zero pad cols
  int pos = (int)(row % LEN_);
  int lvl, H, W, p0;
  if (pos < 4096)      { lvl = 0; H = 64; W = 64; p0 = 0; }
  else if (pos < 5120) { lvl = 1; H = 32; W = 32; p0 = 4096; }
  else                 { lvl = 2; H = 16; W = 16; p0 = 5120; }
  int hw = pos - p0;
  int y = hw / W, x = hw - y * W;
  int cp; float v;
  if (c < 120) { cp = c;       v = (y + 1.0f) / (H + 1e-6f) * 6.283185307179586f; }
  else         { cp = c - 120; v = (x + 1.0f) / (W + 1e-6f) * 6.283185307179586f; }
  float dim_t = powf(10000.f, (float)(2 * (cp / 2)) / 120.f);
  float pv = v / dim_t;
  float pe = (cp & 1) ? cosf(pv) : sinf(pv);
  pe += lvl_emb[lvl * D_ + c];
  float s = src[row * D_ + c];
  qU[idx]   = f2bf(s + pe);
  srcU[idx] = f2bf(s);
}

// Softmax over the 12 (level,point) logits per (row, head)
__global__ void softmax12_k(float* __restrict__ aw) {
  long long i = (long long)blockIdx.x * blockDim.x + threadIdx.x;
  if (i >= (long long)ROWS_ * NH_) return;
  float* p = aw + i * (NL_ * NP_);
  float m = p[0];
#pragma unroll
  for (int j = 1; j < 12; ++j) m = fmaxf(m, p[j]);
  float s = 0.f; float e[12];
#pragma unroll
  for (int j = 0; j < 12; ++j) { e[j] = __expf(p[j] - m); s += e[j]; }
  float inv = 1.f / s;
#pragma unroll
  for (int j = 0; j < 12; ++j) p[j] = e[j] * inv;
}

// ---------------------------------------------------------------------------
// MS-deformable attention sampling: thread per (b, query, head)
// attnU row stride = 256 (pad cols written by h==0 thread)
// ---------------------------------------------------------------------------
__global__ void deform_k(const float* __restrict__ value, const float* __restrict__ off,
                         const float* __restrict__ aw, unsigned short* __restrict__ attnU) {
  long long idx = (long long)blockIdx.x * blockDim.x + threadIdx.x;
  if (idx >= (long long)ROWS_ * NH_) return;
  int h = (int)(idx & 7);
  long long row = idx >> 3;                 // b*LEN_ + q
  int b = (int)(row / LEN_);
  int pos = (int)(row - (long long)b * LEN_);

  int Hq, Wq, p0q;
  if (pos < 4096)      { Hq = 64; Wq = 64; p0q = 0; }
  else if (pos < 5120) { Hq = 32; Wq = 32; p0q = 4096; }
  else                 { Hq = 16; Wq = 16; p0q = 5120; }
  int hw = pos - p0q;
  int yq = hw / Wq, xq = hw - yq * Wq;
  float rx = (xq + 0.5f) / Wq;
  float ry = (yq + 0.5f) / Hq;

  float accv[DH_];
#pragma unroll
  for (int d = 0; d < DH_; ++d) accv[d] = 0.f;

  const int Hs[3] = {64, 32, 16};
  const int P0[3] = {0, 4096, 5120};

  for (int l = 0; l < NL_; ++l) {
    int Hl = Hs[l], Wl = Hs[l], off0 = P0[l];
    for (int p = 0; p < NP_; ++p) {
      int cidx = h * (NL_ * NP_) + l * NP_ + p;
      long long ob = row * (NH_ * NL_ * NP_ * 2) + (long long)cidx * 2;
      float ox = off[ob], oy = off[ob + 1];
      float a = aw[row * (NH_ * NL_ * NP_) + cidx];
      float gx = (rx + ox / Wl) * Wl - 0.5f;
      float gy = (ry + oy / Hl) * Hl - 0.5f;
      float x0f = floorf(gx), y0f = floorf(gy);
      int x0 = (int)x0f, y0 = (int)y0f;
      float wx = gx - x0f, wy = gy - y0f;
#pragma unroll
      for (int cy = 0; cy < 2; ++cy)
#pragma unroll
        for (int cx = 0; cx < 2; ++cx) {
          int xi = x0 + cx, yi = y0 + cy;
          if (xi >= 0 && xi < Wl && yi >= 0 && yi < Hl) {
            float wgt = (cx ? wx : 1.f - wx) * (cy ? wy : 1.f - wy) * a;
            const float* vp =
                value + (((size_t)b * LEN_ + off0 + (size_t)yi * Wl + xi) * NH_ + h) * DH_;
#pragma unroll
            for (int d = 0; d < DH_; ++d) accv[d] += wgt * vp[d];
          }
        }
    }
  }
  unsigned short* o = attnU + row * KP_ + h * DH_;
#pragma unroll
  for (int d = 0; d < DH_; ++d) o[d] = f2bf(accv[d]);
  if (h == 0) {                              // zero pad cols 240..255
    unsigned short* z = attnU + row * KP_ + D_;
#pragma unroll
    for (int j = 0; j < KP_ - D_; ++j) z[j] = 0;
  }
}

// ---------------------------------------------------------------------------
// y = LayerNorm(x + r) * g + b  — one wave32 per row (x,r,yF ld=240; yU ld=256)
// ---------------------------------------------------------------------------
__global__ __launch_bounds__(128)
void addln_k(const float* __restrict__ x, const float* __restrict__ r,
             const float* __restrict__ g, const float* __restrict__ bb,
             float* __restrict__ yF, unsigned short* __restrict__ yU) {
  int row = blockIdx.x * 4 + (threadIdx.x >> 5);
  int lane = threadIdx.x & 31;
  if (row >= ROWS_) return;
  const float* xr = x + (size_t)row * D_;
  const float* rr = r + (size_t)row * D_;
  float vals[8];
  float s = 0.f, s2 = 0.f;
#pragma unroll
  for (int i = 0; i < 8; ++i) {
    int c = lane + i * 32;
    float v = 0.f;
    if (c < D_) { v = xr[c] + rr[c]; s += v; s2 += v * v; }
    vals[i] = v;
  }
#pragma unroll
  for (int o = 16; o > 0; o >>= 1) {
    s  += __shfl_xor(s, o, 32);
    s2 += __shfl_xor(s2, o, 32);
  }
  float mean = s / (float)D_;
  float var  = s2 / (float)D_ - mean * mean;
  float inv  = rsqrtf(var + 1e-5f);
#pragma unroll
  for (int i = 0; i < 8; ++i) {
    int c = lane + i * 32;
    if (c < D_) {
      float yv = (vals[i] - mean) * inv * g[c] + bb[c];
      yF[(size_t)row * D_ + c] = yv;
      if (yU) yU[(size_t)row * KP_ + c] = f2bf(yv);
    } else if (yU) {
      yU[(size_t)row * KP_ + c] = 0;       // zero pad cols
    }
  }
}

// src [B, Len, 240] slice -> NCHW output
__global__ void scatter_k(const float* __restrict__ src, float* __restrict__ out,
                          int H, int W, int p0, long long total) {
  long long idx = (long long)blockIdx.x * blockDim.x + threadIdx.x;
  if (idx >= total) return;
  int HW = H * W;
  int hw = (int)(idx % HW);
  long long t = idx / HW;
  int c = (int)(t % D_);
  int b = (int)(t / D_);
  out[idx] = src[(((size_t)b * LEN_) + p0 + hw) * D_ + c];
}

// ---------------------------------------------------------------------------
extern "C" void kernel_launch(void* const* d_in, const int* in_sizes, int n_in,
                              void* d_out, int out_size, void* d_ws, size_t ws_size,
                              hipStream_t stream) {
  const float* lvl_emb = (const float*)d_in[14];
  const float* so_w = (const float*)d_in[15]; const float* so_b = (const float*)d_in[16];
  const float* aw_w = (const float*)d_in[17]; const float* aw_b = (const float*)d_in[18];
  const float* vp_w = (const float*)d_in[19]; const float* vp_b = (const float*)d_in[20];
  const float* op_w = (const float*)d_in[21]; const float* op_b = (const float*)d_in[22];
  const float* n1_g = (const float*)d_in[23]; const float* n1_b = (const float*)d_in[24];
  const float* ffn1_w = (const float*)d_in[25]; const float* ffn1_b = (const float*)d_in[26];
  const float* ffn2_w = (const float*)d_in[27]; const float* ffn2_b = (const float*)d_in[28];
  const float* n2_g = (const float*)d_in[29]; const float* n2_b = (const float*)d_in[30];

  // Workspace carve
  char* wp = (char*)d_ws;
  auto carve = [&](size_t bytes) -> void* {
    void* p = (void*)wp; wp += (bytes + 255) & ~(size_t)255; return p;
  };
  unsigned short* im2colB = (unsigned short*)carve((size_t)32768 * 1152 * 2); // max level
  unsigned short* wconvB  = (unsigned short*)carve((size_t)256 * 4608 * 2);   // [Np][K]
  float*          srcF    = (float*)carve((size_t)ROWS_ * D_ * 4);
  unsigned short* srcU    = (unsigned short*)carve((size_t)ROWS_ * KP_ * 2);
  unsigned short* qU      = (unsigned short*)carve((size_t)ROWS_ * KP_ * 2);
  float*          offF    = (float*)carve((size_t)ROWS_ * 192 * 4);
  float*          awF     = (float*)carve((size_t)ROWS_ * 96 * 4);
  float*          valF    = (float*)carve((size_t)ROWS_ * D_ * 4);
  unsigned short* attnU   = (unsigned short*)carve((size_t)ROWS_ * KP_ * 2);
  float*          opF     = (float*)carve((size_t)ROWS_ * D_ * 4);
  float*          s2F     = (float*)carve((size_t)ROWS_ * D_ * 4);
  unsigned short* s2U     = (unsigned short*)carve((size_t)ROWS_ * KP_ * 2);
  unsigned short* hU      = (unsigned short*)carve((size_t)ROWS_ * DFF_ * 2);
  float*          f2F     = (float*)carve((size_t)ROWS_ * D_ * 4);
  float*          finF    = (float*)carve((size_t)ROWS_ * D_ * 4);
  unsigned short* wsoU = (unsigned short*)carve((size_t)192 * KP_ * 2);   // [192][256]
  unsigned short* wawU = (unsigned short*)carve((size_t)128 * KP_ * 2);   // [128][256]
  unsigned short* wvpU = (unsigned short*)carve((size_t)256 * KP_ * 2);
  unsigned short* wopU = (unsigned short*)carve((size_t)256 * KP_ * 2);
  unsigned short* wf1U = (unsigned short*)carve((size_t)960 * KP_ * 2);   // [960][256]
  unsigned short* wf2U = (unsigned short*)carve((size_t)256 * DFF_ * 2);  // [256][960]

  auto gemm = [&](const unsigned short* A, const unsigned short* Wt, int M, int N, int Kp,
                  float* oF, unsigned short* oU, const float* bias, const float* scale,
                  int epi, int seg, int ostride, int ooff) {
    dim3 g((N + 63) / 64, (M + 63) / 64);
    hipLaunchKernelGGL(gemm_bf16_wmma, g, dim3(128), 0, stream,
                       A, Wt, M, N, Kp, oF, oU, bias, scale, epi, seg, ostride, ooff);
  };
  auto cvtT = [&](const float* in, unsigned short* Wt, int K, int N, int Kp, int Np) {
    int total = Np * Kp;
    hipLaunchKernelGGL(cvtT_k, dim3((total + 255) / 256), dim3(256), 0, stream,
                       in, Wt, K, N, Kp, total);
  };

  // Passthrough outputs x0, x1
  (void)hipMemcpyAsync(d_out, d_in[0], (size_t)in_sizes[0] * 4,
                       hipMemcpyDeviceToDevice, stream);
  (void)hipMemcpyAsync((char*)d_out + (size_t)in_sizes[0] * 4, d_in[1],
                       (size_t)in_sizes[1] * 4, hipMemcpyDeviceToDevice, stream);

  // Conv projections -> srcF  (implicit GEMM via im2col, BN+ReLU epilogue)
  const float* xs[3] = {(const float*)d_in[2], (const float*)d_in[3], (const float*)d_in[4]};
  const float* cw[3] = {(const float*)d_in[5], (const float*)d_in[8],  (const float*)d_in[11]};
  const float* cg[3] = {(const float*)d_in[6], (const float*)d_in[9],  (const float*)d_in[12]};
  const float* cb[3] = {(const float*)d_in[7], (const float*)d_in[10], (const float*)d_in[13]};
  const int cins[3] = {128, 256, 512};
  const int Hs[3]   = {64, 32, 16};
  const int P0[3]   = {0, 4096, 5120};
  for (int l = 0; l < 3; ++l) {
    int cin = cins[l], H = Hs[l], W = Hs[l];
    int K = cin * 9;            // multiple of 32
    int M = B_ * H * W;         // multiple of 64
    int wtot = 256 * K;         // Wt [Np=256][K]
    hipLaunchKernelGGL(convw_k, dim3((wtot + 255) / 256), dim3(256), 0, stream,
                       cw[l], wconvB, cin, wtot);
    long long itot = (long long)M * K;
    hipLaunchKernelGGL(im2col_k, dim3((unsigned)((itot + 255) / 256)), dim3(256), 0, stream,
                       xs[l], im2colB, cin, H, W, itot);
    gemm(im2colB, wconvB, M, D_, K, srcF, nullptr, cb[l], cg[l],
         /*epi scale_bias_relu*/3, /*seg*/H * W, /*ostride*/LEN_, /*ooff*/P0[l]);
  }

  // pos embedding + q / src bf16 (padded stride 256)
  {
    long long tot = (long long)ROWS_ * KP_;
    hipLaunchKernelGGL(posq_k, dim3((unsigned)((tot + 255) / 256)), dim3(256), 0, stream,
                       srcF, lvl_emb, qU, srcU);
  }

  // Transpose linear weights to bf16 [Np][Kp]
  cvtT(so_w,   wsoU, D_,   192, KP_,  192);
  cvtT(aw_w,   wawU, D_,   96,  KP_,  128);
  cvtT(vp_w,   wvpU, D_,   D_,  KP_,  256);
  cvtT(op_w,   wopU, D_,   D_,  KP_,  256);
  cvtT(ffn1_w, wf1U, D_,   DFF_, KP_, 960);
  cvtT(ffn2_w, wf2U, DFF_, D_,  DFF_, 256);

  // Sampling offsets / attention weights / value projection
  gemm(qU, wsoU, ROWS_, 192, KP_, offF, nullptr, so_b, nullptr, 1, ROWS_, ROWS_, 0);
  gemm(qU, wawU, ROWS_, 96,  KP_, awF,  nullptr, aw_b, nullptr, 1, ROWS_, ROWS_, 0);
  hipLaunchKernelGGL(softmax12_k, dim3((ROWS_ * NH_ + 255) / 256), dim3(256), 0, stream, awF);
  gemm(srcU, wvpU, ROWS_, D_, KP_, valF, nullptr, vp_b, nullptr, 1, ROWS_, ROWS_, 0);

  // Deformable sampling
  hipLaunchKernelGGL(deform_k, dim3((ROWS_ * NH_ + 127) / 128), dim3(128), 0, stream,
                     valF, offF, awF, attnU);

  // Output projection + residual LN1
  gemm(attnU, wopU, ROWS_, D_, KP_, opF, nullptr, op_b, nullptr, 1, ROWS_, ROWS_, 0);
  hipLaunchKernelGGL(addln_k, dim3(ROWS_ / 4), dim3(128), 0, stream,
                     srcF, opF, n1_g, n1_b, s2F, s2U);

  // FFN
  gemm(s2U, wf1U, ROWS_, DFF_, KP_, nullptr, hU, ffn1_b, nullptr, /*bias+relu*/2,
       ROWS_, ROWS_, 0);
  gemm(hU, wf2U, ROWS_, D_, DFF_, f2F, nullptr, ffn2_b, nullptr, 1, ROWS_, ROWS_, 0);
  hipLaunchKernelGGL(addln_k, dim3(ROWS_ / 4), dim3(128), 0, stream,
                     s2F, f2F, n2_g, n2_b, finF, (unsigned short*)nullptr);

  // Scatter back to NCHW outputs
  float* out = (float*)d_out;
  size_t obase = (size_t)in_sizes[0] + (size_t)in_sizes[1];
  for (int l = 0; l < 3; ++l) {
    int H = Hs[l], W = Hs[l];
    long long tot = (long long)B_ * D_ * H * W;
    hipLaunchKernelGGL(scatter_k, dim3((unsigned)((tot + 255) / 256)), dim3(256), 0, stream,
                       finF, out + obase, H, W, P0[l], tot);
    obase += (size_t)tot;
  }
}
